// GraphDecoder_67594195304571
// MI455X (gfx1250) — matrix operand order repaired and checked by reading, not compiled, not run on hardware
//
#include <hip/hip_runtime.h>
#include <math.h>

#define HID 256
#define LATD 128
#define BN 32
#define TN 32
#define EN 2048
#define NBN 10
#define VN 6
#define MROWS (TN * EN + 1)
#define NROWS (BN + TN * EN)

typedef __bf16 bf16_t;
typedef __attribute__((ext_vector_type(16))) __bf16 v16bf;
typedef __attribute__((ext_vector_type(8)))  __bf16 v8bf;
typedef __attribute__((ext_vector_type(8)))  float  v8f;

__device__ __forceinline__ v8f zero_v8f() {
  v8f z;
#pragma unroll
  for (int i = 0; i < 8; ++i) z[i] = 0.f;
  return z;
}

__device__ __forceinline__ v8f wmma_bf16(v16bf a, v16bf b, v8f c) {
  return __builtin_amdgcn_wmma_f32_16x16x32_bf16(false, a, false, b, (short)0, c,
                                                 false, false);
}

// A fragment (16x32 bf16) from a bf16 LDS tile row (row-major).
// Lane half 0 holds K = k0..k0+15; half 1 holds k0+16..k0+31.
// Two 16B ds loads, zero conversions.
__device__ __forceinline__ v16bf load_a_frag_bf(const bf16_t* __restrict__ row,
                                                int k0, int half) {
  v8bf lo = *(const v8bf*)(row + k0 + 16 * half);
  v8bf hi = *(const v8bf*)(row + k0 + 16 * half + 8);
  v16bf a;
#pragma unroll
  for (int e = 0; e < 8; ++e) { a[e] = lo[e]; a[8 + e] = hi[e]; }
  return a;
}

// B fragment (32x16 bf16): weight stored [N][K] row-major bf16; lane (l,half)
// holds column n = l, K = k0+16*half .. +15 (contiguous, 32B aligned).
__device__ __forceinline__ v16bf load_b_frag(const bf16_t* __restrict__ W,
                                             int K, int n, int k0, int half) {
  return *(const v16bf*)(W + (size_t)n * K + k0 + 16 * half);
}

__device__ __forceinline__ v8bf pack8_bf(const float* __restrict__ s) {
  v8bf o;
#pragma unroll
  for (int e = 0; e < 8; ++e) o[e] = (bf16_t)s[e];
  return o;
}

__device__ __forceinline__ float sigmoidf_(float x) {
  return 1.f / (1.f + __expf(-x));
}

// ---------------- weight conversion: f32 -> bf16, split off x-columns -------
__global__ void convert_weights_kernel(
    const float* __restrict__ Wz_w, const float* __restrict__ Wh_w,
    const float* __restrict__ Ur_w, const float* __restrict__ Ui_w,
    const float* __restrict__ W_w,  const float* __restrict__ U_w,
    bf16_t* __restrict__ wz, bf16_t* __restrict__ wh,
    bf16_t* __restrict__ ur, bf16_t* __restrict__ ui,
    bf16_t* __restrict__ w,  bf16_t* __restrict__ u) {
  int i = blockIdx.x * blockDim.x + threadIdx.x;
  int stride = gridDim.x * blockDim.x;
  for (int k = i; k < HID * HID; k += stride) {
    int n = k >> 8, c = k & 255;
    wz[k] = (bf16_t)Wz_w[n * (HID + VN) + VN + c];
    wh[k] = (bf16_t)Wh_w[n * (HID + VN) + VN + c];
    ui[k] = (bf16_t)Ui_w[n * (HID + VN) + VN + c];
    ur[k] = (bf16_t)Ur_w[k];
  }
  for (int k = i; k < HID * (HID + LATD); k += stride) {
    w[k] = (bf16_t)W_w[k];
    u[k] = (bf16_t)U_w[k];
  }
}

// ---------------- one recurrence timestep ----------------------------------
__global__ __launch_bounds__(256) void gru_step_kernel(
    int t, float* __restrict__ Hg, float* __restrict__ curo,
    const int* __restrict__ nei_idx_h, const int* __restrict__ nei_idx_o,
    const int* __restrict__ cur_labels,
    const bf16_t* __restrict__ WzH, const bf16_t* __restrict__ UrB,
    const bf16_t* __restrict__ WhH,
    const float* __restrict__ Wz_w, const float* __restrict__ Wz_b,
    const float* __restrict__ Wr_w, const float* __restrict__ Wr_b,
    const float* __restrict__ Ur_b,
    const float* __restrict__ Wh_w, const float* __restrict__ Wh_b) {
  __shared__ float  sumh[16 * 264];       // f32 (exact epilogue blend)
  __shared__ bf16_t sumhb[16 * 272];      // bf16 copy for the z GEMM
  __shared__ bf16_t hb[2][16 * 272];      // two neighbor tiles, bf16
  __shared__ int    s_idxh[16 * NBN];
  __shared__ int    s_idxo[16 * NBN];
  __shared__ int    s_lbl[16];

  const int tid = threadIdx.x;
  const int e0  = blockIdx.x * 16;
  const int lim = 1 + t * EN;  // rows >= lim are still zero (reference scan)

  if (tid < 16) s_lbl[tid] = cur_labels[t * EN + e0 + tid];
  if (tid < 16 * NBN) {
    int m = tid / NBN, j = tid % NBN;
    int base = ((t * EN) + e0 + m) * NBN + j;
    int ih = nei_idx_h[base];
    int io = nei_idx_o[base];
    s_idxh[tid] = (ih < lim) ? ih : 0;  // row 0 is the zero row
    s_idxo[tid] = (io < lim) ? io : 0;
  }
  __syncthreads();

  // gather: sum_h -> LDS (f32 + bf16), cur_o -> global
  for (int u = tid; u < 16 * 32; u += 256) {
    int m = u >> 5, c8 = u & 31;
    float s[8], o[8];
#pragma unroll
    for (int e = 0; e < 8; ++e) { s[e] = 0.f; o[e] = 0.f; }
    for (int j = 0; j < NBN; ++j) {
      const float* ph = Hg + (size_t)s_idxh[m * NBN + j] * HID + c8 * 8;
      const float* po = Hg + (size_t)s_idxo[m * NBN + j] * HID + c8 * 8;
      float4 a0 = ((const float4*)ph)[0], a1 = ((const float4*)ph)[1];
      float4 b0 = ((const float4*)po)[0], b1 = ((const float4*)po)[1];
      s[0] += a0.x; s[1] += a0.y; s[2] += a0.z; s[3] += a0.w;
      s[4] += a1.x; s[5] += a1.y; s[6] += a1.z; s[7] += a1.w;
      o[0] += b0.x; o[1] += b0.y; o[2] += b0.z; o[3] += b0.w;
      o[4] += b1.x; o[5] += b1.y; o[6] += b1.z; o[7] += b1.w;
    }
    float* sp = sumh + m * 264 + c8 * 8;
    ((float4*)sp)[0] = make_float4(s[0], s[1], s[2], s[3]);
    ((float4*)sp)[1] = make_float4(s[4], s[5], s[6], s[7]);
    *(v8bf*)(sumhb + m * 272 + c8 * 8) = pack8_bf(s);
    float* oc = curo + (size_t)(t * EN + e0 + m) * HID + c8 * 8;
    ((float4*)oc)[0] = make_float4(o[0], o[1], o[2], o[3]);
    ((float4*)oc)[1] = make_float4(o[4], o[5], o[6], o[7]);
  }
  __syncthreads();

  const int lane = tid & 31, wv = tid >> 5;
  const int l = lane & 15, half = lane >> 4;

  // z pre-activation: sum_h @ Wz_h^T
  v8f zacc[2] = {zero_v8f(), zero_v8f()};
  {
    const bf16_t* arow = sumhb + l * 272;
#pragma unroll
    for (int ks = 0; ks < 8; ++ks) {
      v16bf a = load_a_frag_bf(arow, ks * 32, half);
#pragma unroll
      for (int i = 0; i < 2; ++i) {
        v16bf b = load_b_frag(WzH, HID, wv * 32 + i * 16 + l, ks * 32, half);
        zacc[i] = wmma_bf16(a, b, zacc[i]);
      }
    }
  }

  // per-element r bias (constant over neighbors): Wr[:,label] + Wr_b + Ur_b
  float rbias[2][8];
#pragma unroll
  for (int i = 0; i < 2; ++i) {
    int n = wv * 32 + i * 16 + l;
    float wb = Wr_b[n] + Ur_b[n];
#pragma unroll
    for (int r = 0; r < 8; ++r)
      rbias[i][r] = Wr_w[n * VN + s_lbl[r + 8 * half]] + wb;
  }

  float gated[2][8];
#pragma unroll
  for (int i = 0; i < 2; ++i)
#pragma unroll
    for (int r = 0; r < 8; ++r) gated[i][r] = 0.f;

  // neighbors processed in pairs: each Ur B-fragment feeds two WMMAs
  for (int p = 0; p < NBN / 2; ++p) {
    __syncthreads();
    for (int u = tid; u < 2 * 16 * 32; u += 256) {
      int which = u >> 9;
      int v = u & 511;
      int m = v >> 5, c8 = v & 31;
      const float* src =
          Hg + (size_t)s_idxh[m * NBN + 2 * p + which] * HID + c8 * 8;
      float4 x0 = ((const float4*)src)[0];
      float4 x1 = ((const float4*)src)[1];
      float s[8] = {x0.x, x0.y, x0.z, x0.w, x1.x, x1.y, x1.z, x1.w};
      *(v8bf*)(hb[which] + m * 272 + c8 * 8) = pack8_bf(s);
    }
    // prefetch the next pair's rows into cache while this pair computes
    if (p + 1 < NBN / 2 && tid < 64) {
      int m = tid & 15, which = (tid >> 4) & 1;
      __builtin_prefetch(
          Hg + (size_t)s_idxh[m * NBN + 2 * (p + 1) + which] * HID, 0, 1);
      __builtin_prefetch(
          Hg + (size_t)s_idxh[m * NBN + 2 * (p + 1) + which] * HID + 128, 0, 1);
    }
    __syncthreads();

    v8f racc[2][2] = {{zero_v8f(), zero_v8f()}, {zero_v8f(), zero_v8f()}};
    const bf16_t* arow0 = hb[0] + l * 272;
    const bf16_t* arow1 = hb[1] + l * 272;
#pragma unroll
    for (int ks = 0; ks < 8; ++ks) {
      v16bf a0 = load_a_frag_bf(arow0, ks * 32, half);
      v16bf a1 = load_a_frag_bf(arow1, ks * 32, half);
#pragma unroll
      for (int i = 0; i < 2; ++i) {
        v16bf b = load_b_frag(UrB, HID, wv * 32 + i * 16 + l, ks * 32, half);
        racc[0][i] = wmma_bf16(a0, b, racc[0][i]);
        racc[1][i] = wmma_bf16(a1, b, racc[1][i]);
      }
    }
#pragma unroll
    for (int which = 0; which < 2; ++which) {
#pragma unroll
      for (int i = 0; i < 2; ++i) {
        int n = wv * 32 + i * 16 + l;
#pragma unroll
        for (int r = 0; r < 8; ++r) {
          int Mr = r + 8 * half;
          float rv = sigmoidf_(racc[which][i][r] + rbias[i][r]);
          gated[i][r] += rv * (float)hb[which][Mr * 272 + n];
        }
      }
    }
  }

  // write gated to LDS (reuse hb[0]) for the pre_h GEMM
  __syncthreads();
#pragma unroll
  for (int i = 0; i < 2; ++i) {
    int n = wv * 32 + i * 16 + l;
#pragma unroll
    for (int r = 0; r < 8; ++r)
      hb[0][(r + 8 * half) * 272 + n] = (bf16_t)gated[i][r];
  }
  __syncthreads();

  v8f phacc[2] = {zero_v8f(), zero_v8f()};
  {
    const bf16_t* arow = hb[0] + l * 272;
#pragma unroll
    for (int ks = 0; ks < 8; ++ks) {
      v16bf a = load_a_frag_bf(arow, ks * 32, half);
#pragma unroll
      for (int i = 0; i < 2; ++i) {
        v16bf b = load_b_frag(WhH, HID, wv * 32 + i * 16 + l, ks * 32, half);
        phacc[i] = wmma_bf16(a, b, phacc[i]);
      }
    }
  }

#pragma unroll
  for (int i = 0; i < 2; ++i) {
    int n = wv * 32 + i * 16 + l;
    float wzb = Wz_b[n], whb = Wh_b[n];
#pragma unroll
    for (int r = 0; r < 8; ++r) {
      int Mr = r + 8 * half;
      int lbl = s_lbl[Mr];
      float sh = sumh[Mr * 264 + n];
      float z  = sigmoidf_(zacc[i][r] + Wz_w[n * (HID + VN) + lbl] + wzb);
      float ph = tanhf(phacc[i][r] + Wh_w[n * (HID + VN) + lbl] + whb);
      Hg[(size_t)(lim + e0 + Mr) * HID + n] = (1.f - z) * sh + z * ph;
    }
  }
}

// ---------------- root_o = sum_j H_final[root_nei_idx] ---------------------
__global__ void root_o_kernel(const float* __restrict__ Hg,
                              const int* __restrict__ root_nei_idx,
                              float* __restrict__ rooto) {
  int tid0 = blockIdx.x * blockDim.x + threadIdx.x;
  int stride = gridDim.x * blockDim.x;
  for (int u = tid0; u < BN * 64; u += stride) {
    int b = u >> 6, c4 = u & 63;
    float x = 0, y = 0, z = 0, w = 0;
    for (int j = 0; j < NBN; ++j) {
      int idx = root_nei_idx[b * NBN + j];
      float4 a = ((const float4*)(Hg + (size_t)idx * HID))[c4];
      x += a.x; y += a.y; z += a.z; w += a.w;
    }
    ((float4*)(rooto + (size_t)b * HID))[c4] = make_float4(x, y, z, w);
  }
}

// ---------------- word (pred) head -----------------------------------------
__global__ __launch_bounds__(256) void pred_head_kernel(
    const float* __restrict__ Hg, const float* __restrict__ latent,
    const int* __restrict__ contexts, const int* __restrict__ direction,
    const int* __restrict__ pred_targets, const int* __restrict__ root_labels,
    const bf16_t* __restrict__ Wbf, const float* __restrict__ W_b,
    const float* __restrict__ Wo_w, const float* __restrict__ Wo_b,
    float* __restrict__ acc) {
  __shared__ bf16_t abuf[16 * 400];   // [hid(256) | latent(128)] as bf16
  __shared__ float  hbuf[16 * 264];   // layer-1 output (f32 for Wo dot)
  __shared__ float  sc[16 * 8];
  __shared__ int    s_ctx[16], s_tgt[16];
  __shared__ float  s_mask[16];

  const int tid = threadIdx.x;
  const int g0  = blockIdx.x * 16;

  if (tid < 16) {
    int g = g0 + tid;
    int ctx, tgt; float mask;
    if (g < BN) { ctx = g; tgt = root_labels[g]; mask = 1.f; }
    else {
      int q = g - BN;
      ctx = contexts[q]; tgt = pred_targets[q];
      mask = (direction[q] == 1) ? 1.f : 0.f;
    }
    s_ctx[tid] = ctx; s_tgt[tid] = tgt; s_mask[tid] = mask;
  }
  __syncthreads();

  for (int u = tid; u < 16 * 48; u += 256) {
    int m = u / 48, c8 = u % 48;
    int g = g0 + m;
    const float* src;
    if (c8 < 32)  // hidden part: zeros (H row 0) for root rows
      src = Hg + (size_t)((g < BN) ? 0 : (1 + g - BN)) * HID + c8 * 8;
    else
      src = latent + (size_t)s_ctx[m] * LATD + (c8 - 32) * 8;
    float4 x0 = ((const float4*)src)[0];
    float4 x1 = ((const float4*)src)[1];
    float s[8] = {x0.x, x0.y, x0.z, x0.w, x1.x, x1.y, x1.z, x1.w};
    *(v8bf*)(abuf + m * 400 + c8 * 8) = pack8_bf(s);
  }
  __syncthreads();

  const int lane = tid & 31, wv = tid >> 5;
  const int l = lane & 15, half = lane >> 4;

  v8f accv[2] = {zero_v8f(), zero_v8f()};
  {
    const bf16_t* arow = abuf + l * 400;
#pragma unroll
    for (int ks = 0; ks < 12; ++ks) {
      v16bf a = load_a_frag_bf(arow, ks * 32, half);
#pragma unroll
      for (int i = 0; i < 2; ++i) {
        v16bf b = load_b_frag(Wbf, HID + LATD, wv * 32 + i * 16 + l, ks * 32, half);
        accv[i] = wmma_bf16(a, b, accv[i]);
      }
    }
  }
#pragma unroll
  for (int i = 0; i < 2; ++i) {
    int n = wv * 32 + i * 16 + l;
    float bb = W_b[n];
#pragma unroll
    for (int r = 0; r < 8; ++r)
      hbuf[(r + 8 * half) * 264 + n] = fmaxf(accv[i][r] + bb, 0.f);
  }
  __syncthreads();

  if (tid < 16 * VN) {
    int m = tid / VN, v = tid % VN;
    const float* hr = hbuf + m * 264;
    const float* wr = Wo_w + v * HID;
    float s = Wo_b[v];
    for (int k = 0; k < HID; k += 4)
      s += hr[k] * wr[k] + hr[k + 1] * wr[k + 1] + hr[k + 2] * wr[k + 2] +
           hr[k + 3] * wr[k + 3];
    sc[m * 8 + v] = s;
  }
  __syncthreads();

  if (tid < 16) {
    const float* s = sc + tid * 8;
    float mx = s[0]; int am = 0;
#pragma unroll
    for (int v = 1; v < VN; ++v)
      if (s[v] > mx) { mx = s[v]; am = v; }
    float se = 0.f;
#pragma unroll
    for (int v = 0; v < VN; ++v) se += __expf(s[v] - mx);
    float lse = mx + logf(se);
    int tgt = s_tgt[tid];
    float mask = s_mask[tid];
    atomicAdd(&acc[0], (lse - s[tgt]) * mask);
    atomicAdd(&acc[1], ((am == tgt) ? 1.f : 0.f) * mask);
    atomicAdd(&acc[2], mask);
  }
}

// ---------------- stop head ------------------------------------------------
__global__ __launch_bounds__(256) void stop_head_kernel(
    const float* __restrict__ curo, const float* __restrict__ rooto,
    const float* __restrict__ latent,
    const int* __restrict__ cur_labels, const int* __restrict__ contexts,
    const int* __restrict__ direction, const int* __restrict__ root_labels,
    const bf16_t* __restrict__ UiH, const float* __restrict__ Ui_w,
    const float* __restrict__ Ui_b,
    const bf16_t* __restrict__ Ubf, const float* __restrict__ U_b,
    const float* __restrict__ Uo_w, const float* __restrict__ Uo_b,
    float* __restrict__ acc) {
  __shared__ bf16_t obuf[16 * 272];   // cur_o / root_o tile, bf16
  __shared__ bf16_t abuf[16 * 400];   // [h1(256) | latent(128)] as bf16
  __shared__ float  srow[16];
  __shared__ int    s_lbl[16], s_ctx[16];
  __shared__ float  s_tgt[16];

  const int tid = threadIdx.x;
  const int g0  = blockIdx.x * 16;
  const int TE  = TN * EN;

  if (tid < 16) {
    int g = g0 + tid;
    int lbl, ctx; float tgt;
    if (g < TE) { lbl = cur_labels[g]; ctx = contexts[g]; tgt = (float)direction[g]; }
    else { int b = g - TE; lbl = root_labels[b]; ctx = b; tgt = 0.f; }
    s_lbl[tid] = lbl; s_ctx[tid] = ctx; s_tgt[tid] = tgt;
    srow[tid] = 0.f;
  }
  __syncthreads();

  for (int u = tid; u < 16 * 32; u += 256) {
    int m = u >> 5, c8 = u & 31;
    int g = g0 + m;
    const float* op = (g < TE) ? (curo + (size_t)g * HID)
                               : (rooto + (size_t)(g - TE) * HID);
    float4 x0 = ((const float4*)(op + c8 * 8))[0];
    float4 x1 = ((const float4*)(op + c8 * 8))[1];
    float s[8] = {x0.x, x0.y, x0.z, x0.w, x1.x, x1.y, x1.z, x1.w};
    *(v8bf*)(obuf + m * 272 + c8 * 8) = pack8_bf(s);
  }
  for (int u = tid; u < 16 * 16; u += 256) {
    int m = u >> 4, c8 = u & 15;
    const float* src = latent + (size_t)s_ctx[m] * LATD + c8 * 8;
    float4 x0 = ((const float4*)src)[0];
    float4 x1 = ((const float4*)src)[1];
    float s[8] = {x0.x, x0.y, x0.z, x0.w, x1.x, x1.y, x1.z, x1.w};
    *(v8bf*)(abuf + m * 400 + 256 + c8 * 8) = pack8_bf(s);
  }
  __syncthreads();

  const int lane = tid & 31, wv = tid >> 5;
  const int l = lane & 15, half = lane >> 4;

  v8f a1[2] = {zero_v8f(), zero_v8f()};
  {
    const bf16_t* arow = obuf + l * 272;
#pragma unroll
    for (int ks = 0; ks < 8; ++ks) {
      v16bf a = load_a_frag_bf(arow, ks * 32, half);
#pragma unroll
      for (int i = 0; i < 2; ++i) {
        v16bf b = load_b_frag(UiH, HID, wv * 32 + i * 16 + l, ks * 32, half);
        a1[i] = wmma_bf16(a, b, a1[i]);
      }
    }
  }
#pragma unroll
  for (int i = 0; i < 2; ++i) {
    int n = wv * 32 + i * 16 + l;
    float bb = Ui_b[n];
#pragma unroll
    for (int r = 0; r < 8; ++r) {
      int Mr = r + 8 * half;
      abuf[Mr * 400 + n] =
          (bf16_t)fmaxf(a1[i][r] + Ui_w[n * (HID + VN) + s_lbl[Mr]] + bb, 0.f);
    }
  }
  __syncthreads();

  v8f a2[2] = {zero_v8f(), zero_v8f()};
  {
    const bf16_t* arow = abuf + l * 400;
#pragma unroll
    for (int ks = 0; ks < 12; ++ks) {
      v16bf a = load_a_frag_bf(arow, ks * 32, half);
#pragma unroll
      for (int i = 0; i < 2; ++i) {
        v16bf b = load_b_frag(Ubf, HID + LATD, wv * 32 + i * 16 + l, ks * 32, half);
        a2[i] = wmma_bf16(a, b, a2[i]);
      }
    }
  }
  float part[8];
#pragma unroll
  for (int r = 0; r < 8; ++r) part[r] = 0.f;
#pragma unroll
  for (int i = 0; i < 2; ++i) {
    int n = wv * 32 + i * 16 + l;
    float ub = U_b[n], uo = Uo_w[n];
#pragma unroll
    for (int r = 0; r < 8; ++r) part[r] += fmaxf(a2[i][r] + ub, 0.f) * uo;
  }
#pragma unroll
  for (int r = 0; r < 8; ++r) atomicAdd(&srow[r + 8 * half], part[r]);
  __syncthreads();

  if (tid < 16) {
    float s = srow[tid] + Uo_b[0];
    float tgt = s_tgt[tid];
    float loss = fmaxf(s, 0.f) - s * tgt + log1pf(__expf(-fabsf(s)));
    float stop = (s >= 0.f) ? 1.f : 0.f;
    atomicAdd(&acc[3], loss);
    atomicAdd(&acc[4], (stop == tgt) ? 1.f : 0.f);
  }
}

__global__ void finalize_kernel(const float* __restrict__ acc,
                                float* __restrict__ out) {
  if (threadIdx.x == 0 && blockIdx.x == 0) {
    out[0] = acc[0] / (float)BN;       // pred_loss
    out[1] = acc[3] / (float)BN;       // stop_loss
    out[2] = acc[1] / acc[2];          // pred_acc
    out[3] = acc[4] / (float)NROWS;    // stop_acc
  }
}

extern "C" void kernel_launch(void* const* d_in, const int* in_sizes, int n_in,
                              void* d_out, int out_size, void* d_ws,
                              size_t ws_size, hipStream_t stream) {
  (void)in_sizes; (void)n_in; (void)out_size; (void)ws_size;
  const float* latent     = (const float*)d_in[0];
  const int*   nei_idx_h  = (const int*)d_in[1];
  const int*   nei_idx_o  = (const int*)d_in[2];
  const int*   cur_labels = (const int*)d_in[3];
  const int*   contexts   = (const int*)d_in[4];
  const int*   direction  = (const int*)d_in[5];
  const int*   pred_tg    = (const int*)d_in[6];
  const int*   root_lbl   = (const int*)d_in[7];
  const int*   root_nei   = (const int*)d_in[8];
  const float* Wz_w = (const float*)d_in[9];
  const float* Wz_b = (const float*)d_in[10];
  const float* Wr_w = (const float*)d_in[11];
  const float* Wr_b = (const float*)d_in[12];
  const float* Ur_w = (const float*)d_in[13];
  const float* Ur_b = (const float*)d_in[14];
  const float* Wh_w = (const float*)d_in[15];
  const float* Wh_b = (const float*)d_in[16];
  const float* W_w  = (const float*)d_in[17];
  const float* W_b  = (const float*)d_in[18];
  const float* Wo_w = (const float*)d_in[19];
  const float* Wo_b = (const float*)d_in[20];
  const float* U_w  = (const float*)d_in[21];
  const float* U_b  = (const float*)d_in[22];
  const float* Uo_w = (const float*)d_in[23];
  const float* Uo_b = (const float*)d_in[24];
  const float* Ui_w = (const float*)d_in[25];
  const float* Ui_b = (const float*)d_in[26];

  char* ws = (char*)d_ws;
  size_t off = 0;
  auto take = [&](size_t bytes) -> char* {
    char* p = ws + off;
    off = (off + bytes + 255) & ~(size_t)255;
    return p;
  };
  float*  Hg    = (float*)take((size_t)MROWS * HID * sizeof(float));
  float*  curo  = (float*)take((size_t)TN * EN * HID * sizeof(float));
  float*  rooto = (float*)take((size_t)BN * HID * sizeof(float));
  float*  acc   = (float*)take(64);
  bf16_t* WzH   = (bf16_t*)take((size_t)HID * HID * 2);
  bf16_t* WhH   = (bf16_t*)take((size_t)HID * HID * 2);
  bf16_t* UrB   = (bf16_t*)take((size_t)HID * HID * 2);
  bf16_t* UiH   = (bf16_t*)take((size_t)HID * HID * 2);
  bf16_t* Wbf   = (bf16_t*)take((size_t)HID * (HID + LATD) * 2);
  bf16_t* Ubf   = (bf16_t*)take((size_t)HID * (HID + LATD) * 2);

  // Only H row 0 must be zero (gather clamp redirects not-yet-written rows
  // to row 0, matching the reference scan semantics exactly).
  hipMemsetAsync(Hg, 0, HID * sizeof(float), stream);
  hipMemsetAsync(acc, 0, 64, stream);

  convert_weights_kernel<<<64, 256, 0, stream>>>(Wz_w, Wh_w, Ur_w, Ui_w, W_w,
                                                 U_w, WzH, WhH, UrB, UiH, Wbf,
                                                 Ubf);

  for (int t = 0; t < TN; ++t)
    gru_step_kernel<<<EN / 16, 256, 0, stream>>>(
        t, Hg, curo, nei_idx_h, nei_idx_o, cur_labels, WzH, UrB, WhH, Wz_w,
        Wz_b, Wr_w, Wr_b, Ur_b, Wh_w, Wh_b);

  root_o_kernel<<<8, 256, 0, stream>>>(Hg, root_nei, rooto);
  pred_head_kernel<<<NROWS / 16, 256, 0, stream>>>(
      Hg, latent, contexts, direction, pred_tg, root_lbl, Wbf, W_b, Wo_w, Wo_b,
      acc);
  stop_head_kernel<<<NROWS / 16, 256, 0, stream>>>(
      curo, rooto, latent, cur_labels, contexts, direction, root_lbl, UiH,
      Ui_w, Ui_b, Ubf, U_b, Uo_w, Uo_b, acc);
  finalize_kernel<<<1, 1, 0, stream>>>(acc, (float*)d_out);
}
